// TransformerLM_67972152427024
// MI455X (gfx1250) — matrix operand order repaired and checked by reading, not compiled, not run on hardware
//
#include <hip/hip_runtime.h>

// ---------------------------------------------------------------------------
// Problem constants (match reference)
// ---------------------------------------------------------------------------
#define BB 2
#define SS 1024
#define DD 1024
#define HH 16
#define LL 8
#define FF 2752
#define VV 32000
#define DK 64
#define MM (BB * SS)          // 2048 token rows
#define EPSF 1e-5f

// ---------------------------------------------------------------------------
// Types for CDNA5 WMMA
// ---------------------------------------------------------------------------
typedef __bf16 bf16_t;
typedef __attribute__((ext_vector_type(16))) __bf16 v16bf;
typedef __attribute__((ext_vector_type(8)))  __bf16 v8bf;
typedef __attribute__((ext_vector_type(8)))  float  v8f;

__device__ __forceinline__ bf16_t f2bf(float f) {
    // round-to-nearest-even fp32 -> bf16
    unsigned u = __float_as_uint(f);
    u += 0x7FFFu + ((u >> 16) & 1u);
    unsigned short s = (unsigned short)(u >> 16);
    bf16_t b;
    __builtin_memcpy(&b, &s, 2);
    return b;
}

// Async 16-byte global -> LDS copy (gfx1250, ASYNCcnt-tracked, no VGPR data).
// LDS byte offset = low 32 bits of the generic pointer (ISA 10.2 aperture map).
__device__ __forceinline__ void async_b128(unsigned lds_off, const void* gp) {
    asm volatile("global_load_async_to_lds_b128 %0, %1, off"
                 :: "v"(lds_off), "v"((unsigned long long)(uintptr_t)gp)
                 : "memory");
}
__device__ __forceinline__ void wait_async0() {
    asm volatile("s_wait_asynccnt 0x0" ::: "memory");
}

// ---------------------------------------------------------------------------
// Tiled bf16 WMMA GEMM:  C[M,N] = A[M,K](bf16) * W[N,K](bf16)^T (+ fp32 res)
//   Block = 256 threads = 8 waves; tile 128(M) x 64(N), K-step 32.
//   Double-buffered LDS filled by global_load_async_to_lds_b128.
//   Wave (wm 0..3, wn 0..1) owns a 32x32 C sub-tile = 4 WMMAs / K-step.
// ---------------------------------------------------------------------------
#define TM 128
#define TN 64
#define TK 32
#define LDSA 40   // padded bf16 row stride (80 B): keeps 16B chunks aligned

__device__ __forceinline__ void issue_tile(
    bf16_t (*As)[LDSA], bf16_t (*Bs)[LDSA],
    const bf16_t* __restrict__ A, const bf16_t* __restrict__ W,
    int m0, int n0, int K, int k0, int t)
{
    // A tile: 128 rows x 32 bf16 = 512 x 16B chunks, 2 per thread
    #pragma unroll
    for (int cc = 0; cc < 2; ++cc) {
        const int chunk = t + cc * 256;
        const int r = chunk >> 2;
        const int c = (chunk & 3) * 8;
        async_b128((unsigned)(uintptr_t)&As[r][c],
                   &A[(size_t)(m0 + r) * K + k0 + c]);
    }
    // B tile: 64 rows x 32 bf16 = 256 chunks, 1 per thread
    {
        const int r = t >> 2;
        const int c = (t & 3) * 8;
        async_b128((unsigned)(uintptr_t)&Bs[r][c],
                   &W[(size_t)(n0 + r) * K + k0 + c]);
    }
}

__device__ __forceinline__ v16bf frag_a(const bf16_t (*S)[LDSA], int row, int kq) {
    const v8bf lo = *(const v8bf*)&S[row][kq];        // K = kq .. kq+7
    const v8bf hi = *(const v8bf*)&S[row][kq + 16];   // K = kq+16 .. kq+23
    return __builtin_shufflevector(lo, hi, 0,1,2,3,4,5,6,7,8,9,10,11,12,13,14,15);
}
__device__ __forceinline__ v16bf frag_b(const bf16_t (*S)[LDSA], int row, int kb) {
    const v8bf lo = *(const v8bf*)&S[row][kb];        // K = kb .. kb+7
    const v8bf hi = *(const v8bf*)&S[row][kb + 16];   // K = kb+16 .. kb+23
    return __builtin_shufflevector(lo, hi, 0,1,2,3,4,5,6,7,8,9,10,11,12,13,14,15);
}

__global__ __launch_bounds__(256) void gemm_wmma_bf16_kernel(
    const bf16_t* __restrict__ A, const bf16_t* __restrict__ W,
    float* __restrict__ C, const float* __restrict__ res,
    int M, int N, int K)
{
    __shared__ alignas(16) bf16_t As[2][TM][LDSA];
    __shared__ alignas(16) bf16_t Bs[2][TN][LDSA];

    const int t    = threadIdx.x;
    const int lane = t & 31;
    const int wave = t >> 5;
    const int wm   = wave & 3;   // 32-row M sub-tile
    const int wn   = wave >> 2;  // 32-col N sub-tile
    const int m0   = blockIdx.y * TM;
    const int n0   = blockIdx.x * TN;

    v8f acc[2][2];
    #pragma unroll
    for (int i = 0; i < 2; ++i)
        #pragma unroll
        for (int j = 0; j < 2; ++j) acc[i][j] = (v8f){};

    const int nk = K / TK;
    issue_tile(As[0], Bs[0], A, W, m0, n0, K, 0, t);

    const int am = lane & 15;
    const int ak = (lane >> 4) * 8;    // A fragment K-half
    const int bn = lane & 15;
    const int bk = (lane >> 4) * 8;    // B fragment K-half

    for (int i = 0; i < nk; ++i) {
        wait_async0();                 // current buffer landed in LDS
        __syncthreads();
        if (i + 1 < nk)                // prefetch next K-tile into other buffer
            issue_tile(As[(i + 1) & 1], Bs[(i + 1) & 1], A, W, m0, n0, K,
                       (i + 1) * TK, t);

        const int buf = i & 1;
        const v16bf a0 = frag_a(As[buf], wm * 32 + am,      ak);
        const v16bf a1 = frag_a(As[buf], wm * 32 + 16 + am, ak);
        const v16bf b0 = frag_b(Bs[buf], wn * 32 + bn,      bk);
        const v16bf b1 = frag_b(Bs[buf], wn * 32 + 16 + bn, bk);

        acc[0][0] = __builtin_amdgcn_wmma_f32_16x16x32_bf16(
                        false, a0, false, b0, (short)0, acc[0][0], false, false);
        acc[0][1] = __builtin_amdgcn_wmma_f32_16x16x32_bf16(
                        false, a0, false, b1, (short)0, acc[0][1], false, false);
        acc[1][0] = __builtin_amdgcn_wmma_f32_16x16x32_bf16(
                        false, a1, false, b0, (short)0, acc[1][0], false, false);
        acc[1][1] = __builtin_amdgcn_wmma_f32_16x16x32_bf16(
                        false, a1, false, b1, (short)0, acc[1][1], false, false);
        __syncthreads();
    }

    // C/D layout: VGPR r -> lanes 0-15 (M=r, N=lane), lanes 16-31 (M=8+r)
    const int half = lane >> 4;
    const int nn   = lane & 15;
    #pragma unroll
    for (int mi = 0; mi < 2; ++mi) {
        #pragma unroll
        for (int r = 0; r < 8; ++r) {
            const int row = m0 + wm * 32 + mi * 16 + half * 8 + r;
            const size_t off = (size_t)row * N + n0 + wn * 32 + nn;
            float v0 = acc[mi][0][r];
            float v1 = acc[mi][1][r];
            if (res) { v0 += res[off]; v1 += res[off + 16]; }
            C[off]      = v0;
            C[off + 16] = v1;
        }
    }
}

// ---------------------------------------------------------------------------
// fp32 -> bf16 conversion pre-pass (weights & any operand staging)
// One float4 -> packed 4x bf16 per thread; grids are exact.
// ---------------------------------------------------------------------------
__global__ __launch_bounds__(256) void cvt_bf16_kernel(
    const float* __restrict__ src, bf16_t* __restrict__ dst)
{
    const size_t i = (size_t)blockIdx.x * 256 + threadIdx.x;   // float4 slot
    const float4 v = ((const float4*)src)[i];
    union { bf16_t b[4]; uint2 u; } p;
    p.b[0] = f2bf(v.x); p.b[1] = f2bf(v.y);
    p.b[2] = f2bf(v.z); p.b[3] = f2bf(v.w);
    ((uint2*)dst)[i] = p.u;
}

// ---------------------------------------------------------------------------
// Embedding gather: X[row,:] = emb[token[row],:]
// ---------------------------------------------------------------------------
__global__ __launch_bounds__(256) void embed_kernel(
    const int* __restrict__ tok, const float* __restrict__ emb,
    float* __restrict__ X)
{
    const int row = blockIdx.x;
    const int t   = threadIdx.x;
    const int token = tok[row];
    const float4 v = *(const float4*)&emb[(size_t)token * DD + t * 4];
    *(float4*)&X[(size_t)row * DD + t * 4] = v;
}

// ---------------------------------------------------------------------------
// RMSNorm: one block per row; writes bf16 (GEMM A operand)
// ---------------------------------------------------------------------------
__global__ __launch_bounds__(256) void rmsnorm_kernel(
    const float* __restrict__ X, const float* __restrict__ g,
    bf16_t* __restrict__ Y)
{
    __shared__ float red[256];
    const int row = blockIdx.x;
    const int t   = threadIdx.x;
    const float4 xv = *(const float4*)&X[(size_t)row * DD + t * 4];
    red[t] = xv.x * xv.x + xv.y * xv.y + xv.z * xv.z + xv.w * xv.w;
    __syncthreads();
    for (int s = 128; s > 0; s >>= 1) {
        if (t < s) red[t] += red[t + s];
        __syncthreads();
    }
    const float r = 1.0f / sqrtf(red[0] * (1.0f / (float)DD) + EPSF);
    const float4 gv = *(const float4*)&g[t * 4];
    union { bf16_t b[4]; uint2 u; } p;
    p.b[0] = f2bf(xv.x * r * gv.x);
    p.b[1] = f2bf(xv.y * r * gv.y);
    p.b[2] = f2bf(xv.z * r * gv.z);
    p.b[3] = f2bf(xv.w * r * gv.w);
    *(uint2*)&Y[(size_t)row * DD + t * 4] = p.u;
}

// ---------------------------------------------------------------------------
// RoPE in-place on fp32 Q and K. One thread per (b,s,h,freq i), i in 0..31.
// ---------------------------------------------------------------------------
__global__ __launch_bounds__(256) void rope_kernel(
    float* __restrict__ q, float* __restrict__ k)
{
    const int idx = blockIdx.x * 256 + threadIdx.x;   // < B*S*H*32
    const int i   = idx & 31;
    const int hh  = (idx >> 5) & (HH - 1);
    const int row = idx >> 9;                         // b*S+s
    const int s   = row & (SS - 1);
    const float inv = __expf(-((float)(2 * i) / (float)DK) * 9.210340371976184f);
    const float ang = (float)s * inv;
    const float sn = sinf(ang);
    const float cs = cosf(ang);
    const size_t base = (size_t)row * DD + hh * DK + 2 * i;
    const float qe = q[base], qo = q[base + 1];
    q[base]     = qe * cs - qo * sn;
    q[base + 1] = qe * sn + qo * cs;
    const float ke = k[base], ko = k[base + 1];
    k[base]     = ke * cs - ko * sn;
    k[base + 1] = ke * sn + ko * cs;
}

// ---------------------------------------------------------------------------
// Causal flash attention (online softmax). Block = 8 waves = 8 consecutive
// queries of one (b,h); K/V staged in LDS 32 keys at a time.
// Each lane owns 2 of the 64 head dims; output written as bf16.
// ---------------------------------------------------------------------------
__global__ __launch_bounds__(256) void attn_kernel(
    const float* __restrict__ Q, const float* __restrict__ Kt,
    const float* __restrict__ Vt, bf16_t* __restrict__ O)
{
    __shared__ alignas(16) float Qs[8][DK];
    __shared__ alignas(16) float Ks[32][DK + 4];
    __shared__ alignas(16) float Vs[32][DK + 4];

    const int t    = threadIdx.x;
    const int lane = t & 31;
    const int wave = t >> 5;
    const int qb   = blockIdx.x * 8;
    const int h    = blockIdx.y;
    const int b    = blockIdx.z;

    for (int idx = t; idx < 8 * DK; idx += 256) {
        const int r = idx >> 6, c = idx & (DK - 1);
        Qs[r][c] = Q[(size_t)(b * SS + qb + r) * DD + h * DK + c];
    }

    const int q    = qb + wave;
    const int qmax = qb + 7;
    float m = -INFINITY, l = 0.0f, o0 = 0.0f, o1 = 0.0f;
    const int d0 = lane * 2, d1 = lane * 2 + 1;

    for (int kb = 0; kb * 32 <= qmax; ++kb) {
        __syncthreads();
        for (int idx = t; idx < 32 * 16; idx += 256) {      // 32 rows x 16 float4
            const int r = idx >> 4, c4 = (idx & 15) * 4;
            const size_t src = (size_t)(b * SS + kb * 32 + r) * DD + h * DK + c4;
            *(float4*)&Ks[r][c4] = *(const float4*)&Kt[src];
            *(float4*)&Vs[r][c4] = *(const float4*)&Vt[src];
        }
        __syncthreads();

        if (kb * 32 <= q) {                 // uniform per wave
            const int key = kb * 32 + lane;
            float sc;
            if (key <= q) {
                float a = 0.0f;
                for (int dd = 0; dd < DK; ++dd)
                    a += Qs[wave][dd] * Ks[lane][dd];
                sc = a * 0.125f;            // 1/sqrt(64)
            } else {
                sc = -INFINITY;
            }
            float smax = sc;
            for (int off = 16; off; off >>= 1)
                smax = fmaxf(smax, __shfl_xor(smax, off, 32));
            const float mn    = fmaxf(m, smax);
            const float scale = __expf(m - mn);
            const float p     = (key <= q) ? __expf(sc - mn) : 0.0f;
            float ps = p;
            for (int off = 16; off; off >>= 1)
                ps += __shfl_xor(ps, off, 32);
            l  = l * scale + ps;
            o0 *= scale;
            o1 *= scale;
            for (int j = 0; j < 32; ++j) {
                const float pj = __shfl(p, j, 32);
                o0 += pj * Vs[j][d0];
                o1 += pj * Vs[j][d1];
            }
            m = mn;
        }
    }
    const float invl = 1.0f / l;
    union { bf16_t b[2]; unsigned u; } p;
    p.b[0] = f2bf(o0 * invl);
    p.b[1] = f2bf(o1 * invl);
    const size_t ob = (size_t)(b * SS + q) * DD + h * DK;
    *(unsigned*)&O[ob + d0] = p.u;
}

// ---------------------------------------------------------------------------
// SiLU gate: fm = silu(f1) * f3 (fp32 in, bf16 out for the W2 GEMM)
// ---------------------------------------------------------------------------
__global__ __launch_bounds__(256) void silugate_kernel(
    const float* __restrict__ f1, const float* __restrict__ f3,
    bf16_t* __restrict__ fm)
{
    const size_t i = (size_t)blockIdx.x * 256 + threadIdx.x;   // float4 slot
    const float4 a = ((const float4*)f1)[i];
    const float4 g = ((const float4*)f3)[i];
    union { bf16_t b[4]; uint2 u; } p;
    p.b[0] = f2bf(a.x / (1.0f + __expf(-a.x)) * g.x);
    p.b[1] = f2bf(a.y / (1.0f + __expf(-a.y)) * g.y);
    p.b[2] = f2bf(a.z / (1.0f + __expf(-a.z)) * g.z);
    p.b[3] = f2bf(a.w / (1.0f + __expf(-a.w)) * g.w);
    ((uint2*)fm)[i] = p.u;
}

// ---------------------------------------------------------------------------
// Host orchestration
// ---------------------------------------------------------------------------
extern "C" void kernel_launch(void* const* d_in, const int* in_sizes, int n_in,
                              void* d_out, int out_size, void* d_ws, size_t ws_size,
                              hipStream_t stream) {
    (void)in_sizes; (void)n_in; (void)out_size; (void)ws_size;

    const int*   tok   = (const int*)  d_in[0];
    const float* emb   = (const float*)d_in[1];
    const float* Wq    = (const float*)d_in[2];
    const float* Wk    = (const float*)d_in[3];
    const float* Wv    = (const float*)d_in[4];
    const float* Wo    = (const float*)d_in[5];
    const float* W1    = (const float*)d_in[6];
    const float* W2    = (const float*)d_in[7];
    const float* W3    = (const float*)d_in[8];
    const float* g1    = (const float*)d_in[9];
    const float* g2    = (const float*)d_in[10];
    const float* gf    = (const float*)d_in[11];
    const float* Whead = (const float*)d_in[12];
    float* out = (float*)d_out;

    // workspace carving
    const size_t MD = (size_t)MM * DD;      // 2,097,152
    const size_t MF = (size_t)MM * FF;      // 5,636,096
    const size_t VD = (size_t)VV * DD;      // 32,768,000
    char* p = (char*)d_ws;
    float*  x    = (float*)p;  p += MD * sizeof(float);
    float*  q    = (float*)p;  p += MD * sizeof(float);
    float*  k    = (float*)p;  p += MD * sizeof(float);
    float*  v    = (float*)p;  p += MD * sizeof(float);
    float*  f1   = (float*)p;  p += MF * sizeof(float);
    float*  f3   = (float*)p;  p += MF * sizeof(float);
    bf16_t* h    = (bf16_t*)p; p += MD * sizeof(bf16_t);
    bf16_t* att  = (bf16_t*)p; p += MD * sizeof(bf16_t);
    bf16_t* fm   = (bf16_t*)p; p += MF * sizeof(bf16_t);
    bf16_t* wbuf = (bf16_t*)p; p += VD * sizeof(bf16_t);   // total ~164 MB

    auto cvt = [&](const float* src, bf16_t* dst, size_t n) {
        cvt_bf16_kernel<<<(unsigned)(n / 4 / 256), 256, 0, stream>>>(src, dst);
    };
    auto gemm = [&](const bf16_t* A, const bf16_t* W, float* C, const float* res,
                    int M_, int N_, int K_) {
        dim3 grid(N_ / TN, M_ / TM);
        gemm_wmma_bf16_kernel<<<grid, 256, 0, stream>>>(A, W, C, res, M_, N_, K_);
    };

    embed_kernel<<<MM, 256, 0, stream>>>(tok, emb, x);

    const size_t DDsz = (size_t)DD * DD;
    const size_t FDsz = (size_t)FF * DD;

    for (int l = 0; l < LL; ++l) {
        const float* wq = Wq + (size_t)l * DDsz;
        const float* wk = Wk + (size_t)l * DDsz;
        const float* wv = Wv + (size_t)l * DDsz;
        const float* wo = Wo + (size_t)l * DDsz;
        const float* w1 = W1 + (size_t)l * FDsz;
        const float* w2 = W2 + (size_t)l * FDsz;
        const float* w3 = W3 + (size_t)l * FDsz;

        // --- attention block ---
        rmsnorm_kernel<<<MM, 256, 0, stream>>>(x, g1 + (size_t)l * DD, h);
        cvt(wq, wbuf, DDsz);  gemm(h, wbuf, q, nullptr, MM, DD, DD);
        cvt(wk, wbuf, DDsz);  gemm(h, wbuf, k, nullptr, MM, DD, DD);
        cvt(wv, wbuf, DDsz);  gemm(h, wbuf, v, nullptr, MM, DD, DD);
        rope_kernel<<<(BB * SS * HH * 32) / 256, 256, 0, stream>>>(q, k);
        attn_kernel<<<dim3(SS / 8, HH, BB), 256, 0, stream>>>(q, k, v, att);
        cvt(wo, wbuf, DDsz);  gemm(att, wbuf, x, x, MM, DD, DD);   // x += att@Wo^T

        // --- FFN block ---
        rmsnorm_kernel<<<MM, 256, 0, stream>>>(x, g2 + (size_t)l * DD, h);
        cvt(w1, wbuf, FDsz);  gemm(h, wbuf, f1, nullptr, MM, FF, DD);
        cvt(w3, wbuf, FDsz);  gemm(h, wbuf, f3, nullptr, MM, FF, DD);
        silugate_kernel<<<(unsigned)(MF / 4 / 256), 256, 0, stream>>>(f1, f3, fm);
        cvt(w2, wbuf, FDsz);  gemm(fm, wbuf, x, x, MM, DD, FF);    // x += fm@W2^T
    }

    rmsnorm_kernel<<<MM, 256, 0, stream>>>(x, gf, h);
    cvt(Whead, wbuf, VD);
    gemm(h, wbuf, out, nullptr, MM, VV, DD);                       // logits
}
